// AutoRegressiveAdaptiveSpectralConv2d_26834955665750
// MI455X (gfx1250) — matrix operand order repaired and checked by reading, not compiled, not run on hardware
//
#include <hip/hip_runtime.h>
#include <hip/hip_bf16.h>
#include <math.h>

// ---------------- problem constants ----------------
#define BB 4
#define TT 6
#define UU 32
#define HH 256
#define WW 256
#define MX 32
#define MY 32
#define NIMG (BB*TT*UU)      // 768 forward images
#define NOUT (BB*UU)         // 128 inverse images
#define OSCALE_UP 4096.0f    // pre-scale of spectrum before f16 conversion
#define OSCALE_DN (1.0f/(65536.0f*4096.0f)) // 1/(H*W) * 1/OSCALE_UP

typedef __attribute__((ext_vector_type(16))) _Float16 v16h;
typedef __attribute__((ext_vector_type(8)))  _Float16 v8h;
typedef __attribute__((ext_vector_type(8)))  float    v8f;

static __device__ inline v16h make_v16(v8h lo, v8h hi) {
  return __builtin_shufflevector(lo, hi, 0,1,2,3,4,5,6,7,8,9,10,11,12,13,14,15);
}

// A-operand (16x32 f16, MxK): lane L<16 -> row M=L, K = 0..7 (v0..3) and 16..23 (v4..7)
//                             lane L>=16 -> row M=L-16, K = 8..15 and 24..31
__device__ inline v16h load_a_frag(const _Float16* base, int row0, int pitch, int kbase, int lane) {
  const _Float16* p = base + (size_t)(row0 + (lane & 15)) * pitch + kbase + ((lane >> 4) << 3);
  v8h lo = *(const v8h*)(p);
  v8h hi = *(const v8h*)(p + 16);
  return make_v16(lo, hi);
}

// B-operand (32x16 f16, KxN): lane L<16 -> col N=L, K = 0..7 and 16..23;
//                             lane L>=16 -> col N=L-16, K = 8..15 and 24..31.
// `base` must be B^T row-major (row = N, contiguous K), pitch in halves (mult of 8).
__device__ inline v16h load_b_frag(const _Float16* base, int row0, int pitch, int kbase, int lane) {
  const _Float16* p = base + (size_t)(row0 + (lane & 15)) * pitch + kbase + ((lane >> 4) << 3);
  v8h lo = *(const v8h*)(p);
  v8h hi = *(const v8h*)(p + 16);
  return make_v16(lo, hi);
}

__device__ inline v8f wmma16(v16h a, v16h b, v8f c) {
  // D = A*B + C, f16 inputs, f32 accumulate  -> v_wmma_f32_16x16x32_f16
  return __builtin_amdgcn_wmma_f32_16x16x32_f16(false, a, false, b, (short)0, c, false, false);
}

// =====================================================================
// Kernel 0: build f16 DFT tables in workspace.
//  Df   [64][256]  A of fwd stage1: rows 0..31 cos(2πkh/256), rows 32..63 -sin (D = C - iS)
//  Bf2T [64][256]  (B of fwd stage2)^T: row n<32: cos(2πnk/256); n>=32: -sin(2π(n-32)k/256)
//  Pst  [256][64]  A of inv stage1: [h][k]: k<32: cos(2πhk/256); k>=32: sin (P = C + iS)
//  QbT  [512][64]  (B of inv stage2)^T: row n=2w+c interleaved re/im output columns
// =====================================================================
__global__ __launch_bounds__(256) void spectral_tables_kernel(
    _Float16* Df, _Float16* Bf2T, _Float16* Pst, _Float16* QbT) {
  int gid = blockIdx.x * 256 + threadIdx.x;
  const float C0 = 6.28318530717958647692f / 256.0f;
  if (gid < 16384) {                       // Df
    int r = gid >> 8, h = gid & 255, k = r & 31;
    float ang = (float)((k * h) & 255) * C0;
    Df[gid] = (_Float16)((r < 32) ? cosf(ang) : -sinf(ang));
  } else if (gid < 32768) {                // Bf2T
    int idx = gid - 16384;
    int n = idx >> 8, k = idx & 255, nn = n & 31;
    float ang = (float)((nn * k) & 255) * C0;
    Bf2T[idx] = (_Float16)((n < 32) ? cosf(ang) : -sinf(ang));
  } else if (gid < 49152) {                // Pst
    int idx = gid - 32768;
    int h = idx >> 6, k = idx & 63, kk = k & 31;
    float ang = (float)((kk * h) & 255) * C0;
    Pst[idx] = (_Float16)((k < 32) ? cosf(ang) : sinf(ang));
  } else {                                 // QbT
    int idx = gid - 49152;
    int n = idx >> 6, k = idx & 63, w = n >> 1, c = n & 1, kk = k & 31;
    float ang = (float)((kk * w) & 255) * C0;
    float v = (c == 0) ? ((k < 32) ? cosf(ang) : -sinf(ang))
                       : ((k < 32) ? sinf(ang) :  cosf(ang));
    QbT[idx] = (_Float16)v;
  }
}

// =====================================================================
// Kernel 1: forward truncated DFT per image (768 workgroups, 8 waves).
//  Stage 1: Gs(64x256) = Df(64x256) . X(256x256)       [WMMA, X staged f16-transposed in LDS]
//  Stage 2: Fb(64x64)  = Gs(64x256) . Bf2(256x64)      [WMMA, Gs f16 in LDS]
//  Recombine blocks -> F complex, scale by Ws*wt_norm, write f' (interleaved c64).
// =====================================================================
#define XT_PITCH 40     // 32 K + pad, halves (mult of 8)
#define GS_PITCH 264    // 256 K + pad, halves
#define FB_PITCH 68     // floats

__global__ __launch_bounds__(256) void fwd_dft_kernel(
    const float* __restrict__ x, const float* __restrict__ Ws,
    const float* __restrict__ Wt, const _Float16* __restrict__ Df,
    const _Float16* __restrict__ Bf2T, float* __restrict__ fprime) {
  __shared__ __align__(16) char smem[256 * XT_PITCH * 2 + 64 * GS_PITCH * 2];
  _Float16* lds_xt = (_Float16*)smem;                       // live: stage 1
  float*    lds_f  = (float*)smem;                          // live: after stage 2 (overlaps xt)
  _Float16* lds_gs = (_Float16*)(smem + 256 * XT_PITCH * 2);

  const int img  = blockIdx.x;
  const int tid  = threadIdx.x;
  const int lane = tid & 31;
  const int wave = tid >> 5;
  const float* X = x + (size_t)img * (HH * WW);

  // ---- stage 1: 4 M-tiles x 16 N-tiles; wave -> mtile = wave&3, 8 ntiles ----
  const int mt = wave & 3;
  const int ntb = (wave >> 2) * 8;
  v8f acc[8];
  v8f vzero = {0.f,0.f,0.f,0.f,0.f,0.f,0.f,0.f};
#pragma unroll
  for (int j = 0; j < 8; ++j) acc[j] = vzero;

  for (int ks = 0; ks < 8; ++ks) {
    const int kb = ks * 32;
    __syncthreads();   // previous iteration's reads done
    // stage X[kb..kb+31][0..255] transposed+f16 into LDS (coalesced reads)
#pragma unroll 4
    for (int r = 0; r < 32; ++r)
      lds_xt[tid * XT_PITCH + r] = (_Float16)X[(kb + r) * WW + tid];
    __syncthreads();

    // A fragment once, then batch all 16 ds_load_b128 before a dense WMMA burst:
    // the 8 accumulators are independent, so no WMMA->WMMA RAW hazard and the
    // XDL pipe can run back-to-back while DS loads pipeline.
    v16h a = load_a_frag(Df, mt * 16, 256, kb, lane);
    v16h bf[8];
#pragma unroll
    for (int j = 0; j < 8; ++j)
      bf[j] = load_b_frag(lds_xt, (ntb + j) * 16, XT_PITCH, 0, lane);
#pragma unroll
    for (int j = 0; j < 8; ++j)
      acc[j] = wmma16(a, bf[j], acc[j]);
  }
  // park Gs into LDS as f16 (A-operand layout for stage 2)
#pragma unroll
  for (int j = 0; j < 8; ++j) {
    const int col = (ntb + j) * 16 + (lane & 15);
#pragma unroll
    for (int r = 0; r < 8; ++r)
      lds_gs[(mt * 16 + r + 8 * (lane >> 4)) * GS_PITCH + col] = (_Float16)acc[j][r];
  }
  __syncthreads();

  // ---- stage 2: 4x4 tiles; wave -> mtile = wave&3, 2 ntiles ----
  const int mt2 = wave & 3;
  const int nt2 = (wave >> 2) * 2;
  v8f acc2[2] = {vzero, vzero};
  for (int ks = 0; ks < 8; ++ks) {
    const int kb = ks * 32;
    v16h a = load_a_frag(lds_gs, mt2 * 16, GS_PITCH, kb, lane);
    v16h b0 = load_b_frag(Bf2T, (nt2 + 0) * 16, 256, kb, lane);
    v16h b1 = load_b_frag(Bf2T, (nt2 + 1) * 16, 256, kb, lane);
    acc2[0] = wmma16(a, b0, acc2[0]);
    acc2[1] = wmma16(a, b1, acc2[1]);
  }
#pragma unroll
  for (int j = 0; j < 2; ++j) {
    const int col = (nt2 + j) * 16 + (lane & 15);
#pragma unroll
    for (int r = 0; r < 8; ++r)
      lds_f[(mt2 * 16 + r + 8 * (lane >> 4)) * FB_PITCH + col] = acc2[j][r];
  }
  __syncthreads();

  // ---- recombine complex blocks, apply Ws * wt_norm, write f' ----
  float wsum = 0.f;
#pragma unroll
  for (int t = 0; t < TT; ++t) wsum += Wt[t];
  const int t_of_img = (img / UU) % TT;
  const float wtn = Wt[t_of_img] / wsum;

  for (int idx = tid; idx < MX * MY; idx += 256) {
    const int kx = idx >> 5, ky = idx & 31;
    // F = G . (C - iS):  F_re = blk00 - blk11, F_im = blk10 + blk01
    const float re = lds_f[kx * FB_PITCH + ky]        - lds_f[(kx + 32) * FB_PITCH + ky + 32];
    const float im = lds_f[(kx + 32) * FB_PITCH + ky] + lds_f[kx * FB_PITCH + ky + 32];
    const float sc = Ws[idx] * wtn;
    fprime[((size_t)img * (MX * MY) + idx) * 2 + 0] = re * sc;
    fprime[((size_t)img * (MX * MY) + idx) * 2 + 1] = im * sc;
  }
}

// =====================================================================
// Kernel 2: channel mix + temporal sum (scales already folded into f').
//  Per (x,y): Osum[b,o] = sum_{t,i} f'[b,t,i,xy] * R[t,i,o,xy]  (complex)
//  Only ~200 MFLOP -> plain VALU complex FMA; f' cached in LDS, reused by 32 o's.
//  Working set (R 50MB + f' 6.3MB) is resident in the 192MB L2.
// =====================================================================
__global__ __launch_bounds__(128) void mix_kernel(
    const float* __restrict__ fprime, const float* __restrict__ R,
    float* __restrict__ Osum) {
  __shared__ float fl[NIMG * 2];
  const int xy = blockIdx.x;
  const int tid = threadIdx.x;
  for (int idx = tid; idx < NIMG; idx += 128) {
    fl[2 * idx + 0] = fprime[((size_t)idx * (MX * MY) + xy) * 2 + 0];
    fl[2 * idx + 1] = fprime[((size_t)idx * (MX * MY) + xy) * 2 + 1];
  }
  __syncthreads();

  const int b = tid >> 5, oo = tid & 31;
  float ar = 0.f, ai = 0.f;
  for (int t = 0; t < TT; ++t) {
#pragma unroll 4
    for (int i = 0; i < UU; ++i) {
      const int fi_idx = ((b * TT + t) * UU + i) * 2;
      const float fr = fl[fi_idx], fim = fl[fi_idx + 1];
      const float* rp = R + (((size_t)(t * UU + i) * UU + oo) * (MX * MY) + xy) * 2;
      const float rr = rp[0], ri = rp[1];
      ar = fmaf(fr, rr, fmaf(-fim, ri, ar));
      ai = fmaf(fr, ri, fmaf( fim, rr, ai));
    }
  }
  const size_t o_idx = ((size_t)(b * UU + oo) * (MX * MY) + xy) * 2;
  Osum[o_idx + 0] = ar;
  Osum[o_idx + 1] = ai;
}

// =====================================================================
// Kernel 3: zero-padded inverse DFT per (b,o) image (128 workgroups, 8 waves).
//  Stage A: Z(256x64) = Pst(256x64) . Bo(64x64)   Bo = [[O_re, O_im],[-O_im, O_re]]*4096
//  Stage B: Y(256x512) = Z(256x64) . Qbig(64x512) -> interleaved complex rows
// =====================================================================
#define BO_PITCH 72
#define ZL_PITCH 72

__global__ __launch_bounds__(256) void inv_dft_kernel(
    const float* __restrict__ Osum, const _Float16* __restrict__ Pst,
    const _Float16* __restrict__ QbT, float* __restrict__ out) {
  __shared__ __align__(16) _Float16 lds_bo[64 * BO_PITCH];
  __shared__ __align__(16) _Float16 lds_z[256 * ZL_PITCH];

  const int img = blockIdx.x;                 // b*32 + o
  const int tid = threadIdx.x;
  const int lane = tid & 31;
  const int wave = tid >> 5;
  const float* O = Osum + (size_t)img * (MX * MY) * 2;

  // build Bo^T (B-operand layout: row = n, contiguous k), scaled up for f16
  for (int idx = tid; idx < 64 * 64; idx += 256) {
    const int n = idx >> 6, k = idx & 63;
    const int kk = k & 31, nn = n & 31;
    const float* oc = O + (kk * MY + nn) * 2;
    float v;
    if (k < 32) v = (n < 32) ? oc[0] : oc[1];           //  O_re |  O_im
    else        v = (n < 32) ? -oc[1] : oc[0];          // -O_im |  O_re
    lds_bo[n * BO_PITCH + k] = (_Float16)(v * OSCALE_UP);
  }
  __syncthreads();

  // ---- stage A: 16 M-tiles x 4 N-tiles; wave -> Mtiles {2w,2w+1} ----
  // B fragments are shared by both M-tiles: load the 4 of them once per k-step.
  const int mtb = wave * 2;
  v8f vzero = {0.f,0.f,0.f,0.f,0.f,0.f,0.f,0.f};
  v8f za[8];
#pragma unroll
  for (int j = 0; j < 8; ++j) za[j] = vzero;
  for (int ks = 0; ks < 2; ++ks) {
    const int kb = ks * 32;
    v16h bf[4];
#pragma unroll
    for (int nt = 0; nt < 4; ++nt)
      bf[nt] = load_b_frag(lds_bo, nt * 16, BO_PITCH, kb, lane);
#pragma unroll
    for (int m = 0; m < 2; ++m) {
      v16h a = load_a_frag(Pst, (mtb + m) * 16, 64, kb, lane);
#pragma unroll
      for (int nt = 0; nt < 4; ++nt)
        za[m * 4 + nt] = wmma16(a, bf[nt], za[m * 4 + nt]);
    }
  }
#pragma unroll
  for (int m = 0; m < 2; ++m)
#pragma unroll
    for (int nt = 0; nt < 4; ++nt) {
      const int col = nt * 16 + (lane & 15);
#pragma unroll
      for (int r = 0; r < 8; ++r)
        lds_z[((mtb + m) * 16 + r + 8 * (lane >> 4)) * ZL_PITCH + col] =
            (_Float16)za[m * 4 + nt][r];
    }
  __syncthreads();

  // ---- stage B: Y(256x512), wave -> Mtiles {2w,2w+1}, loop 32 N-tiles ----
  for (int m = 0; m < 2; ++m) {
    const int row0 = (mtb + m) * 16;
    v16h a0 = load_a_frag(lds_z, row0, ZL_PITCH, 0, lane);
    v16h a1 = load_a_frag(lds_z, row0, ZL_PITCH, 32, lane);
    for (int nt = 0; nt < 32; ++nt) {
      v16h b0 = load_b_frag(QbT, nt * 16, 64, 0, lane);
      v16h b1 = load_b_frag(QbT, nt * 16, 64, 32, lane);
      v8f acc = vzero;
      acc = wmma16(a0, b0, acc);
      acc = wmma16(a1, b1, acc);
      const int col = nt * 16 + (lane & 15);
#pragma unroll
      for (int r = 0; r < 8; ++r) {
        const int hrow = row0 + r + 8 * (lane >> 4);
        out[((size_t)img * HH + hrow) * (WW * 2) + col] = acc[r] * OSCALE_DN;
      }
    }
  }
}

// =====================================================================
extern "C" void kernel_launch(void* const* d_in, const int* in_sizes, int n_in,
                              void* d_out, int out_size, void* d_ws, size_t ws_size,
                              hipStream_t stream) {
  const float* x  = (const float*)d_in[0];   // [B,T,U,H,W] f32
  const float* R  = (const float*)d_in[1];   // [T,U,U,MX,MY] c64 (interleaved)
  const float* Ws = (const float*)d_in[2];   // [MX,MY] f32
  const float* Wt = (const float*)d_in[3];   // [1,T,1,1,1] f32
  float* out = (float*)d_out;                // [B,1,U,H,W] c64 (interleaved)

  char* w = (char*)d_ws;
  _Float16* Df   = (_Float16*)w; w += 64 * 256 * 2;        // 32 KB
  _Float16* Bf2T = (_Float16*)w; w += 64 * 256 * 2;        // 32 KB
  _Float16* Pst  = (_Float16*)w; w += 256 * 64 * 2;        // 32 KB
  _Float16* QbT  = (_Float16*)w; w += 512 * 64 * 2;        // 64 KB
  float* fprime  = (float*)w;    w += (size_t)NIMG * MX * MY * 2 * 4;  // 6.3 MB
  float* OsumBuf = (float*)w;    w += (size_t)NOUT * MX * MY * 2 * 4;  // 1.05 MB

  spectral_tables_kernel<<<320, 256, 0, stream>>>(Df, Bf2T, Pst, QbT);
  fwd_dft_kernel<<<NIMG, 256, 0, stream>>>(x, Ws, Wt, Df, Bf2T, fprime);
  mix_kernel<<<MX * MY, 128, 0, stream>>>(fprime, R, OsumBuf);
  inv_dft_kernel<<<NOUT, 256, 0, stream>>>(OsumBuf, Pst, QbT, out);
}
